// GQAKernel_63702954934390
// MI455X (gfx1250) — matrix operand order repaired and checked by reading, not compiled, not run on hardware
//
#include <hip/hip_runtime.h>
#include <hip/hip_bf16.h>
#include <math.h>

// ---------------------------------------------------------------------------
// GQA causal flash-attention forward for MI455X (gfx1250, wave32, WMMA).
// Pass 1: convert K/V fp32 -> f16 once (V pre-transposed) into workspace.
// Pass 2: flash attention; double-buffered f16 tiles stream global->LDS via
//         CDNA5 async loads (ASYNCcnt) overlapped with 16x
//         v_wmma_f32_16x16x32_f16 per 64-key tile; softmax in exp2 domain
//         with raw v_exp_f32.
// ---------------------------------------------------------------------------

typedef __attribute__((ext_vector_type(16))) _Float16 v16h;
typedef __attribute__((ext_vector_type(8)))  float    v8f;

#define S_LEN 4096
#define NH    8
#define NKV   2
#define DH    64
// 1/sqrt(64) folded with log2(e): softmax runs in exp2 domain.
#define QK_SCALE_LOG2 (0.125f * 1.44269504088896f)

#define WAVES 8
#define QTW   16                    // queries per wave (WMMA M)
#define QTB   (QTW * WAVES)         // 128 queries per block
#define KTILE 64                    // keys per iteration (4 WMMA N-groups)

// LDS row strides in halfs (rows are 160B = multiple of 32B -> aligned v16h).
#define KSTRIDE 80                  // [key][d]   : 64 dims + 16 pad
#define VSTRIDE 80                  // [d][key]   : 64 keys + 16 pad
#define KBYTES  (KTILE * KSTRIDE * 2)
#define VBYTES  (DH * VSTRIDE * 2)

// ---------------------------------------------------------------------------
// Pass 1: K -> f16 [hkv][s][d], V -> f16 transposed [hkv][d][s]
// ---------------------------------------------------------------------------
__global__ __launch_bounds__(256)
void cvt_kv_kernel(const float* __restrict__ k, const float* __restrict__ v,
                   _Float16* __restrict__ kh, _Float16* __restrict__ vt)
{
    const int idx = blockIdx.x * 256 + threadIdx.x;   // over S*NKV*DH
    if (idx >= S_LEN * NKV * DH) return;
    const int d   = idx & (DH - 1);
    const int sh  = idx >> 6;
    const int hkv = sh & (NKV - 1);
    const int s   = sh >> 1;
    kh[((size_t)hkv * S_LEN + s) * DH + d] = (_Float16)k[idx];
    vt[((size_t)hkv * DH + d) * S_LEN + s] = (_Float16)v[idx];
}

// ---------------------------------------------------------------------------
// CDNA5 async global->LDS copy (16B per lane), tracked with ASYNCcnt.
// ---------------------------------------------------------------------------
__device__ __forceinline__ void async_b128(unsigned lds_off,
                                           unsigned long long gbase,
                                           unsigned voff)
{
    asm volatile("global_load_async_to_lds_b128 %0, %1, %2"
                 :: "v"(lds_off), "v"(voff), "s"(gbase) : "memory");
}
__device__ __forceinline__ void wait_async_le4()
{
    asm volatile("s_wait_asynccnt 0x4" ::: "memory");
}
__device__ __forceinline__ void wait_async_le0()
{
    asm volatile("s_wait_asynccnt 0x0" ::: "memory");
}

// ---------------------------------------------------------------------------
// Pass 2: flash attention forward
// ---------------------------------------------------------------------------
__global__ __launch_bounds__(256)
void gqa_fa_fwd_kernel(const float* __restrict__ q,
                       const _Float16* __restrict__ kh,
                       const _Float16* __restrict__ vt,
                       float* __restrict__ out)
{
    __shared__ __attribute__((aligned(32))) _Float16 Klds[2][KTILE * KSTRIDE];
    __shared__ __attribute__((aligned(32))) _Float16 Vlds[2][DH * VSTRIDE];
    __shared__ __attribute__((aligned(32))) _Float16 Plds[WAVES * 16 * KTILE];

    const int tid  = threadIdx.x;
    const int wave = tid >> 5;
    const int lane = tid & 31;
    const int hi   = lane >> 4;         // lane half -> fragment K-range
    const int ln   = lane & 15;
    const int h    = blockIdx.y;
    const int hkv  = h >> 2;            // 4 query heads per kv head
    const int qb   = blockIdx.x * QTB;
    const int qw   = qb + wave * QTW;   // wave's first query row

    const unsigned klds0 = (unsigned)(uintptr_t)&Klds[0][0];
    const unsigned vlds0 = (unsigned)(uintptr_t)&Vlds[0][0];
    const unsigned long long kgb =
        (unsigned long long)(uintptr_t)(kh + (size_t)hkv * S_LEN * DH);
    const unsigned long long vgb =
        (unsigned long long)(uintptr_t)(vt + (size_t)hkv * DH * S_LEN);

    // Per-thread async-chunk geometry (2 chunks for K, 2 for V per tile).
    const unsigned c0 = (unsigned)tid, c1 = (unsigned)tid + 256;
    const unsigned row0 = c0 >> 3, part0 = (c0 & 7) * 16;
    const unsigned row1 = c1 >> 3, part1 = (c1 & 7) * 16;

    // ---- Q A-fragments (16x32 f16 layout), scaled into exp2 domain --------
    // lane holds row m = ln; half j -> dim = 16*hi + j.
    v16h aQ0, aQ1;
    {
        const float* qrow = q + ((size_t)(qw + ln) * NH + h) * DH;
#pragma unroll
        for (int i = 0; i < 8; ++i) {
            aQ0[i]     = (_Float16)(qrow[     16*hi + i] * QK_SCALE_LOG2);
            aQ0[8 + i] = (_Float16)(qrow[ 8 + 16*hi + i] * QK_SCALE_LOG2);
            aQ1[i]     = (_Float16)(qrow[32 + 16*hi + i] * QK_SCALE_LOG2);
            aQ1[8 + i] = (_Float16)(qrow[40 + 16*hi + i] * QK_SCALE_LOG2);
        }
    }

    v8f o[4] = {{}, {}, {}, {}};        // 16x64 fp32 accumulator
    float mrow[8], lrow[8];
#pragma unroll
    for (int r = 0; r < 8; ++r) { mrow[r] = -INFINITY; lrow[r] = 0.0f; }

    const int qmax = qw + QTW - 1;
    const int kend = qb + QTB;
    _Float16* pw = &Plds[wave * 16 * KTILE];

    // ---- pipelined tile loop: issue t+1, wait t, barrier, compute t --------
    // Issue tile0 into buffer 0 before the loop.
    {
        async_b128(klds0 + row0 * (KSTRIDE * 2) + part0, kgb,
                   row0 * (DH * 2) + part0);
        async_b128(vlds0 + row0 * (VSTRIDE * 2) + part0, vgb,
                   row0 * (S_LEN * 2) + part0);
        async_b128(klds0 + row1 * (KSTRIDE * 2) + part1, kgb,
                   row1 * (DH * 2) + part1);
        async_b128(vlds0 + row1 * (VSTRIDE * 2) + part1, vgb,
                   row1 * (S_LEN * 2) + part1);
    }

    for (int j0 = 0; j0 < kend; j0 += KTILE) {
        const unsigned cur = ((unsigned)j0 >> 6) & 1;
        if (j0 + KTILE < kend) {
            const unsigned nxt = cur ^ 1;
            const unsigned jn  = (unsigned)(j0 + KTILE);
            async_b128(klds0 + nxt * KBYTES + row0 * (KSTRIDE * 2) + part0, kgb,
                       (jn + row0) * (DH * 2) + part0);
            async_b128(vlds0 + nxt * VBYTES + row0 * (VSTRIDE * 2) + part0, vgb,
                       row0 * (S_LEN * 2) + jn * 2 + part0);
            async_b128(klds0 + nxt * KBYTES + row1 * (KSTRIDE * 2) + part1, kgb,
                       (jn + row1) * (DH * 2) + part1);
            async_b128(vlds0 + nxt * VBYTES + row1 * (VSTRIDE * 2) + part1, vgb,
                       row1 * (S_LEN * 2) + jn * 2 + part1);
            wait_async_le4();               // retire tile t, keep t+1 in flight
        } else {
            wait_async_le0();
        }
        __syncthreads();                    // tile t visible to all waves

        if (j0 <= qmax) {
            const _Float16* Kc = &Klds[cur][0];
            const _Float16* Vc = &Vlds[cur][0];

            // ---- S = Q*K^T : 8x WMMA (4 key groups x 2 dim chunks) ---------
            v8f sc[4];
#pragma unroll
            for (int kg = 0; kg < 4; ++kg) {
                const v16h b0 = *(const v16h*)&Kc[(16*kg + ln) * KSTRIDE      + 16*hi];
                const v16h b1 = *(const v16h*)&Kc[(16*kg + ln) * KSTRIDE + 32 + 16*hi];
                v8f z = {};
                z = __builtin_amdgcn_wmma_f32_16x16x32_f16(false, aQ0, false, b0, (short)0, z, false, false);
                sc[kg] = __builtin_amdgcn_wmma_f32_16x16x32_f16(false, aQ1, false, b1, (short)0, z, false, false);
            }

            // ---- causal mask: only the wave's diagonal tile needs it -------
            if (j0 + (KTILE - 1) > qw) {          // wave-uniform branch
#pragma unroll
                for (int kg = 0; kg < 4; ++kg) {
                    const int key = j0 + 16*kg + ln;
#pragma unroll
                    for (int r = 0; r < 8; ++r) {
                        const int qi = qw + r + 8*hi;
                        sc[kg][r] = (key <= qi) ? sc[kg][r] : -INFINITY;
                    }
                }
            }

            // ---- online softmax, exp2 domain (row m = r+8*hi, n = ln) ------
            float alpha[8];
#pragma unroll
            for (int r = 0; r < 8; ++r) {
                float tmax = fmaxf(fmaxf(sc[0][r], sc[1][r]),
                                   fmaxf(sc[2][r], sc[3][r]));
#pragma unroll
                for (int off = 8; off > 0; off >>= 1)
                    tmax = fmaxf(tmax, __shfl_xor(tmax, off, 16));
                const float mnew = fmaxf(mrow[r], tmax);
                const float a = __builtin_amdgcn_exp2f(mrow[r] - mnew);
                float p0 = __builtin_amdgcn_exp2f(sc[0][r] - mnew);
                float p1 = __builtin_amdgcn_exp2f(sc[1][r] - mnew);
                float p2 = __builtin_amdgcn_exp2f(sc[2][r] - mnew);
                float p3 = __builtin_amdgcn_exp2f(sc[3][r] - mnew);
                float tsum = (p0 + p1) + (p2 + p3);
#pragma unroll
                for (int off = 8; off > 0; off >>= 1)
                    tsum += __shfl_xor(tsum, off, 16);
                lrow[r] = lrow[r] * a + tsum;
                mrow[r] = mnew;
                alpha[r] = a;
                sc[0][r] = p0; sc[1][r] = p1; sc[2][r] = p2; sc[3][r] = p3;
            }
#pragma unroll
            for (int nc = 0; nc < 4; ++nc)
#pragma unroll
                for (int r = 0; r < 8; ++r)
                    o[nc][r] *= alpha[r];

            // ---- P: store pre-swizzled so aP loads are single v16h reads ---
            // dest(m, key=16*kg+ln) = m*64 + 16*kg + ln
#pragma unroll
            for (int kg = 0; kg < 4; ++kg) {
                const int sub = 16*kg + ln;
#pragma unroll
                for (int r = 0; r < 8; ++r)
                    pw[(r + 8*hi) * KTILE + sub] = (_Float16)sc[kg][r];
            }
            asm volatile("s_wait_dscnt 0" ::: "memory");   // same-wave LDS RAW
            const v16h aP0 = *(const v16h*)&pw[ln * KTILE +      16*hi];
            const v16h aP1 = *(const v16h*)&pw[ln * KTILE + 32 + 16*hi];

            // ---- O += P*V : 8x WMMA (4 col groups x 2 key chunks) ----------
#pragma unroll
            for (int nc = 0; nc < 4; ++nc) {
                const v16h bv0 = *(const v16h*)&Vc[(16*nc + ln) * VSTRIDE      + 16*hi];
                const v16h bv1 = *(const v16h*)&Vc[(16*nc + ln) * VSTRIDE + 32 + 16*hi];
                o[nc] = __builtin_amdgcn_wmma_f32_16x16x32_f16(false, aP0, false, bv0, (short)0, o[nc], false, false);
                o[nc] = __builtin_amdgcn_wmma_f32_16x16x32_f16(false, aP1, false, bv1, (short)0, o[nc], false, false);
            }
        }
        __syncthreads();                    // protect buffer reuse next iter
    }

    // ---- epilogue: normalize and store fp32 --------------------------------
#pragma unroll
    for (int r = 0; r < 8; ++r) {
        const float inv = __builtin_amdgcn_rcpf(lrow[r]);
        const int m = r + 8*hi;
        float* orow = out + ((size_t)(qw + m) * NH + h) * DH;
        orow[     ln] = o[0][r] * inv;
        orow[16 + ln] = o[1][r] * inv;
        orow[32 + ln] = o[2][r] * inv;
        orow[48 + ln] = o[3][r] * inv;
    }
}

extern "C" void kernel_launch(void* const* d_in, const int* in_sizes, int n_in,
                              void* d_out, int out_size, void* d_ws, size_t ws_size,
                              hipStream_t stream) {
    (void)in_sizes; (void)n_in; (void)out_size; (void)ws_size;
    const float* q = (const float*)d_in[0];
    const float* k = (const float*)d_in[1];
    const float* v = (const float*)d_in[2];
    float* out = (float*)d_out;

    _Float16* kh = (_Float16*)d_ws;                          // [2][4096][64] f16
    _Float16* vt = kh + (size_t)NKV * S_LEN * DH;            // [2][64][4096] f16

    const int total = S_LEN * NKV * DH;
    cvt_kv_kernel<<<(total + 255) / 256, 256, 0, stream>>>(k, v, kh, vt);

    dim3 grid(S_LEN / QTB, NH, 1);
    gqa_fa_fwd_kernel<<<grid, 256, 0, stream>>>(q, kh, vt, out);
}
